// Sparse_Graph_Attention_Layer_55937654063759
// MI455X (gfx1250) — matrix operand order repaired and checked by reading, not compiled, not run on hardware
//
#include <hip/hip_runtime.h>
#include <hip/hip_bf16.h>

typedef __attribute__((ext_vector_type(16))) _Float16 v16h;
typedef __attribute__((ext_vector_type(8)))  float    v8f;

#define D_MODEL 128
#define OUT_DIM 64
#define DEG     16
#define ALPHA_LRELU 0.2f

// ---------------------------------------------------------------------------
// Kernel A: wh = x @ W via v_wmma_f32_16x16x32_f16 (f32 accumulate),
// fused with per-node attention scalars s1 = wh . a_src, s2 = wh . a_dst.
// One wave32 per 16-row tile of x. W is staged transposed in LDS as f16.
// ---------------------------------------------------------------------------
__global__ __launch_bounds__(256) void gat_gemm_wmma(
    const float* __restrict__ x, const float* __restrict__ W,
    const float* __restrict__ a_values,
    float* __restrict__ wh, float* __restrict__ s1, float* __restrict__ s2,
    int n_nodes, int n_tiles)
{
  // Wt[n][k] : transposed W, f16, 64*128*2B = 16 KB of the 320 KB WGP LDS
  __shared__ __align__(32) _Float16 Wt[OUT_DIM * D_MODEL];
  for (int i = threadIdx.x; i < D_MODEL * OUT_DIM; i += 256) {
    int k = i >> 6;          // row of W   (0..127)
    int n = i & 63;          // col of W   (0..63)
    Wt[n * D_MODEL + k] = (_Float16)W[i];
  }
  __syncthreads();

  const int tile = (blockIdx.x * 256 + threadIdx.x) >> 5;   // global wave id
  if (tile >= n_tiles) return;                              // wave-uniform
  const int lane = threadIdx.x & 31;
  const int h    = lane >> 4;    // half-wave selector
  const int j    = lane & 15;

  const int row0 = tile * 16;
  int arow = row0 + j;                     // A-frag: lanes 0-15 / 16-31 -> M = lane%16
  if (arow >= n_nodes) arow = n_nodes - 1; // clamp (n divisible by 16 anyway)
  const float* __restrict__ xr = x + (size_t)arow * D_MODEL;

  v8f c[4];
  #pragma unroll
  for (int t = 0; t < 4; ++t)
    #pragma unroll
    for (int e = 0; e < 8; ++e) c[t][e] = 0.0f;

  #pragma unroll
  for (int k0 = 0; k0 < D_MODEL; k0 += 32) {
    // A fragment: lane half h holds K = k0+16h..k0+16h+15 (elems 0-15),
    // matching the B fragment's contiguous K order.
    v16h a;
    {
      const float* p = xr + k0 + h * 16;
      float4 u0 = *(const float4*)(p);
      float4 u1 = *(const float4*)(p + 4);
      float4 u2 = *(const float4*)(p + 8);
      float4 u3 = *(const float4*)(p + 12);
      a[0] = (_Float16)u0.x; a[1] = (_Float16)u0.y; a[2] = (_Float16)u0.z; a[3] = (_Float16)u0.w;
      a[4] = (_Float16)u1.x; a[5] = (_Float16)u1.y; a[6] = (_Float16)u1.z; a[7] = (_Float16)u1.w;
      a[8] = (_Float16)u2.x; a[9] = (_Float16)u2.y; a[10]= (_Float16)u2.z; a[11]= (_Float16)u2.w;
      a[12]= (_Float16)u3.x; a[13]= (_Float16)u3.y; a[14]= (_Float16)u3.z; a[15]= (_Float16)u3.w;
    }
    #pragma unroll
    for (int t = 0; t < 4; ++t) {
      // B fragment: lane holds N = t*16 + j, K = k0 + 16h .. k0 + 16h + 15
      // (two ds_load_b128 from LDS, 32B aligned)
      const v16h b = *(const v16h*)&Wt[(t * 16 + j) * D_MODEL + k0 + h * 16];
      c[t] = __builtin_amdgcn_wmma_f32_16x16x32_f16(
          /*neg_a=*/false, a, /*neg_b=*/false, b,
          /*c_mod=*/(short)0, c[t], /*reuse_a=*/false, /*reuse_b=*/false);
    }
  }

  // a_src / a_dst coefficients for this lane's column within each N-tile
  float asrc[4], adst[4];
  #pragma unroll
  for (int t = 0; t < 4; ++t) {
    asrc[t] = a_values[t * 16 + j];
    adst[t] = a_values[OUT_DIM + t * 16 + j];
  }

  // D layout: lane l, acc elem r -> (m = 8h + r, n = t*16 + j)
  float p1[8], p2[8];
  #pragma unroll
  for (int r = 0; r < 8; ++r) {
    const int row = row0 + h * 8 + r;
    float pa = 0.0f, pd = 0.0f;
    #pragma unroll
    for (int t = 0; t < 4; ++t) {
      float v = c[t][r];
      if (row < n_nodes) wh[(size_t)row * OUT_DIM + t * 16 + j] = v;
      pa += v * asrc[t];
      pd += v * adst[t];
    }
    p1[r] = pa; p2[r] = pd;
  }

  // reduce over the 16 columns within each half-wave
  #pragma unroll
  for (int r = 0; r < 8; ++r) {
    #pragma unroll
    for (int m = 8; m >= 1; m >>= 1) {
      p1[r] += __shfl_xor(p1[r], m, 16);
      p2[r] += __shfl_xor(p2[r], m, 16);
    }
  }
  if (j < 8) {
    float v1 = 0.0f, v2 = 0.0f;
    #pragma unroll
    for (int r = 0; r < 8; ++r) if (r == j) { v1 = p1[r]; v2 = p2[r]; }
    const int row = row0 + h * 8 + j;
    if (row < n_nodes) { s1[row] = v1; s2[row] = v2; }
  }
}

// ---------------------------------------------------------------------------
// Kernel B: per-node edge softmax-style aggregation.
// src = repeat(arange(N), 16) => edges [16*i, 16*i+16) belong to node i.
// One wave32 per node; lanes 0-15 own one edge each, all 32 lanes accumulate
// two coalesced columns of the gathered wh rows (L2-resident gather).
// ---------------------------------------------------------------------------
__global__ __launch_bounds__(256) void gat_edge_agg(
    const float* __restrict__ wh, const float* __restrict__ s1,
    const float* __restrict__ s2, const int* __restrict__ dst,
    float* __restrict__ out, int n_nodes)
{
  const int node = (blockIdx.x * 256 + threadIdx.x) >> 5;
  if (node >= n_nodes) return;
  const int lane = threadIdx.x & 31;

  // edge weights on lanes 0..15
  float w  = 0.0f;
  int   de = 0;
  if (lane < DEG) {
    de = dst[node * DEG + lane];
    float sc = s1[node] + s2[de];
    float lr = sc > 0.0f ? sc : ALPHA_LRELU * sc;
    w = __expf(-lr);
  }
  // row_sum (upper 16 lanes contribute 0)
  float rs = w;
  #pragma unroll
  for (int m = 16; m >= 1; m >>= 1) rs += __shfl_xor(rs, m, 32);

  float acc0 = 0.0f, acc1 = 0.0f;
  #pragma unroll
  for (int e = 0; e < DEG; ++e) {
    const int   dd = __shfl(de, e, 32);
    const float we = __shfl(w,  e, 32);
    const float* __restrict__ row = wh + (size_t)dd * OUT_DIM;
    acc0 += we * row[lane];        // cols  0..31  (128B coalesced)
    acc1 += we * row[lane + 32];   // cols 32..63
  }

  const float inv = 1.0f / rs;
  float o0 = acc0 * inv;
  float o1 = acc1 * inv;
  o0 = o0 > 0.0f ? o0 : (__expf(o0) - 1.0f);   // elu, alpha=1
  o1 = o1 > 0.0f ? o1 : (__expf(o1) - 1.0f);
  out[(size_t)node * OUT_DIM + lane]      = o0;
  out[(size_t)node * OUT_DIM + lane + 32] = o1;
}

// ---------------------------------------------------------------------------
extern "C" void kernel_launch(void* const* d_in, const int* in_sizes, int n_in,
                              void* d_out, int out_size, void* d_ws, size_t ws_size,
                              hipStream_t stream) {
  const float* x  = (const float*)d_in[0];   // (N, 128) f32
  const float* W  = (const float*)d_in[1];   // (128, 64) f32
  const float* av = (const float*)d_in[2];   // (1, 128) f32
  const int*   ei = (const int*)d_in[3];     // (2, E) int

  float* out = (float*)d_out;

  const int n_nodes = in_sizes[0] / D_MODEL;
  const int n_edges = in_sizes[3] / 2;
  const int* dst = ei + n_edges;             // row 1 of edge_index

  // workspace: wh (N*64 f32) | s1 (N f32) | s2 (N f32)
  float* wh = (float*)d_ws;
  float* s1 = wh + (size_t)n_nodes * OUT_DIM;
  float* s2 = s1 + n_nodes;

  const int n_tiles = (n_nodes + 15) / 16;
  const int blocksA = (n_tiles * 32 + 255) / 256;
  gat_gemm_wmma<<<blocksA, 256, 0, stream>>>(x, W, av, wh, s1, s2, n_nodes, n_tiles);

  const int blocksB = (n_nodes * 32 + 255) / 256;
  gat_edge_agg<<<blocksB, 256, 0, stream>>>(wh, s1, s2, dst, out, n_nodes);

  (void)n_in; (void)out_size; (void)ws_size;
}
